// FeatureNet_57174604644985
// MI455X (gfx1250) — matrix operand order repaired and checked by reading, not compiled, not run on hardware
//
#include <hip/hip_runtime.h>

typedef _Float16 v16h __attribute__((ext_vector_type(16)));
typedef _Float16 v8h  __attribute__((ext_vector_type(8)));
typedef float    v8f  __attribute__((ext_vector_type(8)));

#define B_TOT   128
#define T_LEN   200
#define C_CODES 40
#define NCODE   2000
#define D_AUX   16
#define E_DIM   128
#define FOUR_E  512
#define HMLP    256
#define ROWS    16     // batch rows per workgroup
#define NTILES  32     // 512 / 16
#define KSTEPS  4      // 128 / 32
#define HSTRIDE 136    // halves per row in x16/h16 (pad for bank-conflict-free A loads)
#define ZSTRIDE 514    // floats per row in z buffer
#define NTHREADS 512

// ---- LDS layout (bytes), total 317312 < 320KB/WGP ----
constexpr int OFF_WXF   = 0;                         // 128KB f16 B-fragments of W_x
constexpr int OFF_WHF   = OFF_WXF   + 128 * 1024;    // 128KB f16 B-fragments of W_h
constexpr int OFF_Z     = OFF_WHF   + 128 * 1024;    // 16 x 514 f32 gate pre-activations
constexpr int OFF_X16   = OFF_Z     + ZSTRIDE * 16 * 4;
constexpr int OFF_H16   = OFF_X16   + HSTRIDE * 16 * 2;
constexpr int OFF_LAST  = OFF_H16   + HSTRIDE * 16 * 2;   // 16 x 128 f32 last hidden
constexpr int OFF_CODE  = OFF_LAST  + E_DIM * 16 * 4;     // 16 x 40 int codes
constexpr int OFF_KEEP  = OFF_CODE  + C_CODES * 16 * 4;   // 16 x 2 u32 keep bitmask
constexpr int OFF_BLSTM = OFF_KEEP  + 32 * 4;
constexpr int OFF_BLIN  = OFF_BLSTM + FOUR_E * 4;
constexpr int OFF_LEN   = OFF_BLIN  + E_DIM * 4;
constexpr int OFF_SSQ   = OFF_LEN   + 16 * 4;
constexpr int SMEM_TOTAL = OFF_SSQ  + 16 * 4;

union HV { v16h v; v8h h[2]; };

__device__ __forceinline__ float fast_sigmoid(float x) {
  x = fminf(fmaxf(x, -30.f), 30.f);
  return 1.f / (1.f + __expf(-x));
}
__device__ __forceinline__ float fast_tanh(float x) {
  x = fminf(fmaxf(x, -15.f), 15.f);
  float e = __expf(-2.f * x);
  return (1.f - e) / (1.f + e);
}

// A fragment (16x32 f16) from a row-major padded [16][HSTRIDE] f16 LDS tile.
// lane<16 : halves 0-7 = K 32k..32k+7, halves 8-15 = K 32k+16..32k+23
// lane>=16: halves 0-7 = K 32k+8..32k+15, halves 8-15 = K 32k+24..32k+31
__device__ __forceinline__ v16h load_afrag(const _Float16* base, int lane, int k) {
  int row = lane & 15;
  int kb  = 32 * k + ((lane >> 4) << 3);
  HV u;
  u.h[0] = *(const v8h*)(base + row * HSTRIDE + kb);
  u.h[1] = *(const v8h*)(base + row * HSTRIDE + kb + 16);
  return u.v;
}

// B fragment: pre-swizzled, 512 contiguous halves per fragment, 16 per lane.
__device__ __forceinline__ v16h load_bfrag(const _Float16* fragbase, int lane) {
  HV u;
  u.h[0] = *(const v8h*)(fragbase + lane * 16);
  u.h[1] = *(const v8h*)(fragbase + lane * 16 + 8);
  return u.v;
}

__global__ __launch_bounds__(NTHREADS, 1)
void featurenet_fused(const int* __restrict__ code,
                      const float* __restrict__ aux,
                      const int* __restrict__ length,
                      const float* __restrict__ W_lin,
                      const float* __restrict__ b_lin,
                      const float* __restrict__ W_x,
                      const float* __restrict__ W_h,
                      const float* __restrict__ b_lstm,
                      const float* __restrict__ W0,
                      const float* __restrict__ b0v,
                      const float* __restrict__ W1,
                      const float* __restrict__ b1v,
                      float* __restrict__ out)
{
  extern __shared__ __align__(16) unsigned char smem[];
  _Float16*     wxf     = (_Float16*)    (smem + OFF_WXF);
  _Float16*     whf     = (_Float16*)    (smem + OFF_WHF);
  float*        zbuf    = (float*)       (smem + OFF_Z);
  _Float16*     x16     = (_Float16*)    (smem + OFF_X16);
  _Float16*     h16     = (_Float16*)    (smem + OFF_H16);
  float*        lastb   = (float*)       (smem + OFF_LAST);
  int*          codes_s = (int*)         (smem + OFF_CODE);
  unsigned int* keepw   = (unsigned int*)(smem + OFF_KEEP);
  float*        blstm_s = (float*)       (smem + OFF_BLSTM);
  float*        blin_s  = (float*)       (smem + OFF_BLIN);
  int*          len_s   = (int*)         (smem + OFF_LEN);
  float*        ssq_s   = (float*)       (smem + OFF_SSQ);

  const int tid   = threadIdx.x;
  const int lane  = tid & 31;
  const int wave  = tid >> 5;     // 0..15, one wave per batch row in the frontend
  const int wg    = blockIdx.x;
  const int brow0 = wg * ROWS;

  // ---------------- setup: convert + swizzle weights into LDS B-fragments ----
  for (int idx = tid; idx < E_DIM * FOUR_E; idx += NTHREADS) {
    int within = idx & 511;
    int frag   = idx >> 9;          // k*32 + nt
    int l      = within >> 4;       // lane slot 0..31
    int h      = within & 15;       // half slot 0..15
    int nt     = frag & 31;
    int k      = frag >> 5;
    int col    = nt * 16 + (l & 15);
    int kk     = k * 32 + ((l >> 4) << 3) + (h & 7) + ((h >> 3) << 4);   // lane<16: K..K+7,K+16..K+23; lane>=16: +8
    wxf[idx] = (_Float16)W_x[kk * FOUR_E + col];
    whf[idx] = (_Float16)W_h[kk * FOUR_E + col];
  }
  for (int idx = tid; idx < ROWS * HSTRIDE; idx += NTHREADS) h16[idx] = (_Float16)0.f;
  if (tid < FOUR_E) blstm_s[tid] = b_lstm[tid];
  if (tid < E_DIM)  blin_s[tid]  = b_lin[tid];
  if (tid < ROWS)   len_s[tid]   = length[brow0 + tid];
  __syncthreads();

  float creg[4] = {0.f, 0.f, 0.f, 0.f};
  const int erow  = tid & 127;   // element column owned in gate stage
  const int rbase = tid >> 7;    // 0..3 -> rows rbase, rbase+4, +8, +12

  for (int t = 0; t < T_LEN; ++t) {
    // (a) stage codes for this timestep; clear keep masks
    if (tid < 32) keepw[tid] = 0u;
    for (int i = tid; i < ROWS * C_CODES; i += NTHREADS) {
      int r = i / C_CODES, c = i - r * C_CODES;
      codes_s[i] = code[((size_t)(brow0 + r) * T_LEN + t) * C_CODES + c];
    }
    __syncthreads();

    // (b) dedup (multi-hot .set semantics): keep only first occurrence of nonzero code
    for (int i = tid; i < ROWS * C_CODES; i += NTHREADS) {
      int r = i / C_CODES, c = i - r * C_CODES;
      int j = codes_s[i];
      if (j != 0) {
        bool dup = false;
        const int* rowc = codes_s + r * C_CODES;
        for (int c2 = 0; c2 < c; ++c2) dup = dup || (rowc[c2] == j);
        if (!dup) atomicOr(&keepw[2 * r + (c >> 5)], 1u << (c & 31));
      }
    }
    __syncthreads();

    // (c) sparse frontend: wave w builds x[row=w][0:128] = relu(gather + aux*W + b)
    {
      int r = wave;
      float a0 = blin_s[lane], a1 = blin_s[lane + 32];
      float a2 = blin_s[lane + 64], a3 = blin_s[lane + 96];
      unsigned int kw0 = keepw[2 * r], kw1 = keepw[2 * r + 1];
      const int* rowc = codes_s + r * C_CODES;
      for (int c = 0; c < C_CODES; ++c) {
        unsigned int bit = (c < 32) ? ((kw0 >> c) & 1u) : ((kw1 >> (c - 32)) & 1u);
        if (bit) {  // wave-uniform branch
          const float* wrow = W_lin + (size_t)(rowc[c] - 1) * E_DIM;
          a0 += wrow[lane];      a1 += wrow[lane + 32];
          a2 += wrow[lane + 64]; a3 += wrow[lane + 96];
        }
      }
      const float* arow = aux + ((size_t)(brow0 + r) * T_LEN + t) * D_AUX;
      for (int k = 0; k < D_AUX; ++k) {
        float av = arow[k];
        const float* wrow = W_lin + (size_t)(NCODE + k) * E_DIM;
        a0 = fmaf(av, wrow[lane],      a0);
        a1 = fmaf(av, wrow[lane + 32], a1);
        a2 = fmaf(av, wrow[lane + 64], a2);
        a3 = fmaf(av, wrow[lane + 96], a3);
      }
      _Float16* xr = x16 + r * HSTRIDE;
      xr[lane]      = (_Float16)fmaxf(a0, 0.f);
      xr[lane + 32] = (_Float16)fmaxf(a1, 0.f);
      xr[lane + 64] = (_Float16)fmaxf(a2, 0.f);
      xr[lane + 96] = (_Float16)fmaxf(a3, 0.f);
    }
    __syncthreads();

    // (d) z = x@W_x + h@W_h  via WMMA; each wave owns N-tiles {2w, 2w+1}
    {
      v16h ax[KSTEPS], ah[KSTEPS];
      #pragma unroll
      for (int k = 0; k < KSTEPS; ++k) {
        ax[k] = load_afrag(x16, lane, k);
        ah[k] = load_afrag(h16, lane, k);
      }
      #pragma unroll
      for (int q = 0; q < 2; ++q) {
        int nt = 2 * wave + q;
        v8f acc = {0.f, 0.f, 0.f, 0.f, 0.f, 0.f, 0.f, 0.f};
        #pragma unroll
        for (int k = 0; k < KSTEPS; ++k) {
          v16h bx = load_bfrag(wxf + ((k * NTILES + nt) << 9), lane);
          acc = __builtin_amdgcn_wmma_f32_16x16x32_f16(false, ax[k], false, bx,
                                                       (short)0, acc, false, false);
        }
        #pragma unroll
        for (int k = 0; k < KSTEPS; ++k) {
          v16h bh = load_bfrag(whf + ((k * NTILES + nt) << 9), lane);
          acc = __builtin_amdgcn_wmma_f32_16x16x32_f16(false, ah[k], false, bh,
                                                       (short)0, acc, false, false);
        }
        int colb = nt * 16 + (lane & 15);
        int roff = (lane < 16) ? 0 : 8;
        #pragma unroll
        for (int r = 0; r < 8; ++r)
          zbuf[(r + roff) * ZSTRIDE + colb] = acc[r];
      }
    }
    __syncthreads();

    // (e) gates (keras order i,f,g,o), state update, snapshot at t==len-1
    #pragma unroll
    for (int rr = 0; rr < 4; ++rr) {
      int r = rbase + 4 * rr;
      const float* zr = zbuf + r * ZSTRIDE;
      float zi = zr[erow]       + blstm_s[erow];
      float zf = zr[erow + 128] + blstm_s[erow + 128];
      float zg = zr[erow + 256] + blstm_s[erow + 256];
      float zo = zr[erow + 384] + blstm_s[erow + 384];
      float cn = fast_sigmoid(zf) * creg[rr] + fast_sigmoid(zi) * fast_tanh(zg);
      creg[rr] = cn;
      float hn = fast_sigmoid(zo) * fast_tanh(cn);
      h16[r * HSTRIDE + erow] = (_Float16)hn;
      if (t == len_s[r] - 1) lastb[r * E_DIM + erow] = hn;
    }
    __syncthreads();
  }

  // ---------------- MLP head + L2 normalize (tiny, VALU) ----------------
  {
    int n  = tid & 255;
    int r0 = tid >> 8;   // 0..1
    #pragma unroll
    for (int j = 0; j < 8; ++j) {
      int r = r0 + 2 * j;
      float acc = b0v[n];
      const float* lr = lastb + r * E_DIM;
      for (int k = 0; k < E_DIM; ++k) acc = fmaf(lr[k], W0[k * HMLP + n], acc);
      zbuf[r * 260 + n] = fmaxf(acc, 0.f);   // reuse z buffer as y0[16][260]
    }
  }
  __syncthreads();
  if (tid < ROWS) ssq_s[tid] = 0.f;
  __syncthreads();
  float yv[8];
  {
    int n  = tid & 255;
    int r0 = tid >> 8;
    #pragma unroll
    for (int j = 0; j < 8; ++j) {
      int r = r0 + 2 * j;
      float acc = b1v[n];
      const float* yr = zbuf + r * 260;
      for (int k = 0; k < HMLP; ++k) acc = fmaf(yr[k], W1[k * HMLP + n], acc);
      yv[j] = acc;
      atomicAdd(&ssq_s[r], acc * acc);
    }
  }
  __syncthreads();
  {
    int n  = tid & 255;
    int r0 = tid >> 8;
    #pragma unroll
    for (int j = 0; j < 8; ++j) {
      int r = r0 + 2 * j;
      out[(size_t)(brow0 + r) * HMLP + n] = yv[j] * rsqrtf(ssq_s[r]);
    }
  }
}

extern "C" void kernel_launch(void* const* d_in, const int* in_sizes, int n_in,
                              void* d_out, int out_size, void* d_ws, size_t ws_size,
                              hipStream_t stream) {
  (void)in_sizes; (void)n_in; (void)out_size; (void)d_ws; (void)ws_size;
  const int*   code    = (const int*)  d_in[0];
  const float* aux     = (const float*)d_in[1];
  const int*   length  = (const int*)  d_in[2];
  // d_in[3] = is_training (unused at inference)
  const float* W_lin   = (const float*)d_in[4];
  const float* b_lin   = (const float*)d_in[5];
  const float* W_x     = (const float*)d_in[6];
  const float* W_h     = (const float*)d_in[7];
  const float* b_lstm  = (const float*)d_in[8];
  const float* W0      = (const float*)d_in[9];
  const float* b0v     = (const float*)d_in[10];
  const float* W1      = (const float*)d_in[11];
  const float* b1v     = (const float*)d_in[12];
  float* out = (float*)d_out;

  dim3 grid(B_TOT / ROWS);   // 8 workgroups, 16 batch rows each
  dim3 block(NTHREADS);      // 16 waves of 32
  featurenet_fused<<<grid, block, SMEM_TOTAL, stream>>>(
      code, aux, length, W_lin, b_lin, W_x, W_h, b_lstm, W0, b0v, W1, b1v, out);
}
